// MoERouterSwitch_19825569038531
// MI455X (gfx1250) — compile-verified, matrix operand order checked by reading, not run on hardware
//
#include <hip/hip_runtime.h>
#include <hip/hip_bf16.h>

// ---------------------------------------------------------------------------
// MoE router (Switch): logits = x@W + b ; softmax ; top-3 mask ; column sums.
// N=32768, D=2048, E=64. Memory-bound (~290MB @ 23.3TB/s ~ 12.5us), so the
// GEMM runs on v_wmma_f32_16x16x32_bf16 (f32->bf16 cvt in VALU) to keep the
// matrix work far below the HBM floor. wave32 / WGP-aware tiling.
// Round 3: two M-tiles per wave -> each LDS B-fragment feeds two back-to-back
// WMMAs (ds:wmma 1:1 instead of 2:1); logits scratch unioned with the W^T
// tile to keep LDS at ~33KB.
// ---------------------------------------------------------------------------

typedef __bf16 bf16x8  __attribute__((ext_vector_type(8)));
typedef __bf16 bf16x16 __attribute__((ext_vector_type(16)));
typedef float  v8f     __attribute__((ext_vector_type(8)));

#define D_MODEL 2048
#define N_EXP   64
#define KC      128          // K-chunk staged per iteration (4 WMMA K-steps)
#define BM      256          // rows per workgroup (8 waves x 2 tiles x 16 rows)
#define WTS     136          // LDS stride (bf16) for W^T tile: 128+8 pad -> bank-rotated
#define LSTR    65           // LDS stride (f32) for logits: 64+1 pad

// A fragment: 16-bit A 16x32 layout; lanes 0-15 take K{0..7,16..23},
// lanes 16-31 take K{8..15,24..31} of their row (khi pre-applied by caller).
__device__ __forceinline__ bf16x16 load_afrag(const float* __restrict__ ap) {
    const float4 f0 = *(const float4*)(ap);
    const float4 f1 = *(const float4*)(ap + 4);
    const float4 f2 = *(const float4*)(ap + 16);
    const float4 f3 = *(const float4*)(ap + 20);
    const float tf[16] = { f0.x, f0.y, f0.z, f0.w, f1.x, f1.y, f1.z, f1.w,
                           f2.x, f2.y, f2.z, f2.w, f3.x, f3.y, f3.z, f3.w };
    bf16x16 a;
#pragma unroll
    for (int i = 0; i < 16; ++i) a[i] = (__bf16)tf[i];
    return a;
}

// B fragment (mirror layout) from LDS W^T; conflict-free via WTS pad.
__device__ __forceinline__ bf16x16 load_bfrag(const __bf16* wp) {
    const bf16x8 lo = *(const bf16x8*)(wp);
    const bf16x8 hi = *(const bf16x8*)(wp + 16);
    bf16x16 b;
#pragma unroll
    for (int i = 0; i < 8; ++i) { b[i] = lo[i]; b[i + 8] = hi[i]; }
    return b;
}

__global__ __launch_bounds__(256)
void moe_router_gemm_kernel(const float* __restrict__ x,
                            const float* __restrict__ W,
                            const float* __restrict__ b,
                            float* __restrict__ expert_mask,
                            float* __restrict__ route_prob)
{
    // W^T tile (GEMM phase) and logits scratch (epilogue phase) never live
    // simultaneously -> union keeps total LDS ~33KB for WGP co-residency.
    __shared__ union {
        __bf16 wt[N_EXP * WTS];                  // [expert][k], bf16
        float  logits[8 * 16 * LSTR];            // per-wave 16x64 (+pad)
    } sU;
    __shared__ float sBias[N_EXP];

    const int tid  = threadIdx.x;
    const int wave = tid >> 5;
    const int lane = tid & 31;
    const int lrow = lane & 15;                  // row-in-tile / col-in-tile
    const int khi  = (lane >> 4) ? 8 : 0;        // upper half-lanes take K+8 slice

    if (tid < N_EXP) sBias[tid] = b[tid];

    const int r0 = blockIdx.x * BM + wave * 32;             // wave's first row
    const float* xrow0 = x + (size_t)(r0 + lrow) * D_MODEL; // M-tile 0 row
    const float* xrow1 = xrow0 + (size_t)16 * D_MODEL;      // M-tile 1 row

    v8f acc[2][4];
#pragma unroll
    for (int m = 0; m < 2; ++m)
#pragma unroll
        for (int e = 0; e < 4; ++e) acc[m][e] = (v8f)0.0f;

    for (int k0 = 0; k0 < D_MODEL; k0 += KC) {
        // ---- stage W[k0:k0+KC, :] -> LDS as bf16 W^T (coalesced f32 reads) ----
        __syncthreads();                                   // prev chunk consumed
        const float* Wc = W + (size_t)k0 * N_EXP;
#pragma unroll
        for (int i = tid; i < KC * N_EXP; i += 256) {
            const int k = i >> 6;                          // i / 64
            const int e = i & 63;                          // i % 64
            sU.wt[e * WTS + k] = (__bf16)Wc[i];
        }
        // prefetch next chunks while we compute this one (global_prefetch_b8)
        if (k0 + KC < D_MODEL) {
            __builtin_prefetch(Wc + (size_t)KC * N_EXP + (tid << 2), 0, 0);
            __builtin_prefetch(xrow0 + k0 + KC + khi, 0, 0);
            __builtin_prefetch(xrow1 + k0 + KC + khi, 0, 0);
        }
        __syncthreads();                                   // W^T tile visible

        // ---- 4 K-steps; each B fragment feeds two back-to-back WMMAs ----
#pragma unroll
        for (int t = 0; t < 4; ++t) {
            const int koff = k0 + t * 32 + khi;
            const bf16x16 a0 = load_afrag(xrow0 + koff);
            const bf16x16 a1 = load_afrag(xrow1 + koff);
#pragma unroll
            for (int e = 0; e < 4; ++e) {
                const bf16x16 bf =
                    load_bfrag(&sU.wt[(e * 16 + lrow) * WTS + t * 32 + khi]);
                acc[0][e] = __builtin_amdgcn_wmma_f32_16x16x32_bf16(
                    false, a0, false, bf, (short)0, acc[0][e], false, false);
                acc[1][e] = __builtin_amdgcn_wmma_f32_16x16x32_bf16(
                    false, a1, false, bf, (short)0, acc[1][e], false, false);
            }
        }
    }

    // ---- epilogue in two batches (one per M-tile); logits scratch aliases
    //      the W^T tile, so the first barrier also retires all sWT readers ----
    float* lw = &sU.logits[wave * 16 * LSTR];
    const int rbase = (lane < 16) ? 0 : 8;
    for (int m = 0; m < 2; ++m) {
        __syncthreads();
        // C/D layout: VGPR j = row j (lanes 0-15) or row j+8 (lanes 16-31),
        // col = lane%16 + 16*etile
#pragma unroll
        for (int e = 0; e < 4; ++e) {
            const int col = lrow + e * 16;
#pragma unroll
            for (int j = 0; j < 8; ++j) lw[(rbase + j) * LSTR + col] = acc[m][e][j];
        }
        __syncthreads();

        if (lane < 16) {
            const int grow = r0 + m * 16 + lane;
            const float* lr = &lw[lane * LSTR];

            // pass 1: max (== top-1), smallest index on ties (matches top_k)
            float v1 = -3.402823466e+38f; int i1 = 0;
#pragma unroll 8
            for (int e = 0; e < N_EXP; ++e) {
                const float v = lr[e] + sBias[e];
                if (v > v1) { v1 = v; i1 = e; }
            }
            // pass 2: exp-sum + top-2
            float s = 0.0f, v2 = -3.402823466e+38f; int i2 = 0;
#pragma unroll 8
            for (int e = 0; e < N_EXP; ++e) {
                const float v = lr[e] + sBias[e];
                s += __expf(v - v1);
                if (e != i1 && v > v2) { v2 = v; i2 = e; }
            }
            // pass 3: top-3
            float v3 = -3.402823466e+38f; int i3 = 0;
#pragma unroll 8
            for (int e = 0; e < N_EXP; ++e) {
                const float v = lr[e] + sBias[e];
                if (e != i1 && e != i2 && v > v3) { v3 = v; i3 = e; }
            }
            // pass 4: write route_prob and mask rows
            const float inv_s = __frcp_rn(s);
            float* rp = route_prob + (size_t)grow * N_EXP;
            float* mk = expert_mask + (size_t)grow * N_EXP;
#pragma unroll 8
            for (int e = 0; e < N_EXP; ++e) {
                const float v = lr[e] + sBias[e];
                rp[e] = __expf(v - v1) * inv_s;
                mk[e] = (e == i1 || e == i2 || e == i3) ? 1.0f : 0.0f;
            }
        }
    }
}

// Deterministic column reduction: importance[e] = load[e] = sum_n route_prob[n][e].
// route_prob (8 MB) is L2-resident (192 MB L2), so strided reads are cheap.
__global__ __launch_bounds__(256)
void moe_router_reduce_kernel(const float* __restrict__ route_prob,
                              float* __restrict__ importance,
                              float* __restrict__ load_out, int N)
{
    __shared__ float s[256];
    const int e = blockIdx.x;
    float sum = 0.0f;
    for (int n = threadIdx.x; n < N; n += 256)
        sum += route_prob[(size_t)n * N_EXP + e];
    s[threadIdx.x] = sum;
    __syncthreads();
#pragma unroll
    for (int off = 128; off > 0; off >>= 1) {
        if (threadIdx.x < off) s[threadIdx.x] += s[threadIdx.x + off];
        __syncthreads();
    }
    if (threadIdx.x == 0) { importance[e] = s[0]; load_out[e] = s[0]; }
}

extern "C" void kernel_launch(void* const* d_in, const int* in_sizes, int n_in,
                              void* d_out, int out_size, void* d_ws, size_t ws_size,
                              hipStream_t stream) {
    const float* x = (const float*)d_in[0];
    const float* W = (const float*)d_in[1];
    const float* b = (const float*)d_in[2];
    const int N = in_sizes[0] / D_MODEL;             // 32768

    float* expert_mask = (float*)d_out;
    float* route_prob  = expert_mask + (size_t)N * N_EXP;
    float* importance  = route_prob  + (size_t)N * N_EXP;
    float* load_out    = importance  + N_EXP;

    moe_router_gemm_kernel<<<N / BM, 256, 0, stream>>>(x, W, b, expert_mask, route_prob);
    moe_router_reduce_kernel<<<N_EXP, 256, 0, stream>>>(route_prob, importance, load_out, N);
}
